// Net_39719857553901
// MI455X (gfx1250) — matrix operand order repaired and checked by reading, not compiled, not run on hardware
//
#include <hip/hip_runtime.h>
#include <hip/hip_bf16.h>

// ---------------------------------------------------------------------------
// CDNA5 / gfx1250: conv + fc layers as implicit GEMM on
// v_wmma_f32_16x16x32_bf16 (wave32, one 16x16 tile per wave).
//  - im2col k->offset decode via LDS-resident lookup table.
//  - fc GEMMs stage A-tiles into double-buffered LDS with
//    global_load_async_to_lds_b32 + s_wait_asynccnt (builtin-guarded),
//    else coalesced global load + ds_store fallback.
// ---------------------------------------------------------------------------

typedef __attribute__((ext_vector_type(16))) __bf16 v16bf;
typedef __attribute__((ext_vector_type(8)))  float  v8f;

union FragAB {
    v16bf          v;
    unsigned short u[16];
    unsigned int   d[8];
};

#if defined(__gfx1250__) && __has_builtin(__builtin_amdgcn_global_load_async_to_lds_b32) && \
    __has_builtin(__builtin_amdgcn_s_wait_asynccnt)
#define USE_ASYNC_LDS 1
#else
#define USE_ASYNC_LDS 0
#endif

typedef __attribute__((address_space(1))) int as1_int;   // global ("__device__ int*")
typedef __attribute__((address_space(3))) int as3_int;   // LDS    ("__shared__ int*")

__device__ __forceinline__ unsigned short f2bf(float f) {
    unsigned int x = __float_as_uint(f);
    x += 0x7fffu + ((x >> 16) & 1u);          // round-to-nearest-even
    return (unsigned short)(x >> 16);
}
__device__ __forceinline__ float bf2f(unsigned short u) {
    return __uint_as_float(((unsigned int)u) << 16);
}

// --------------------------- utility kernels -------------------------------

__global__ void k_f32_to_bf16(const float* __restrict__ src,
                              unsigned short* __restrict__ dst, long long n) {
    long long i = (long long)blockIdx.x * blockDim.x + threadIdx.x;
    long long stride = (long long)gridDim.x * blockDim.x;
    for (; i < n; i += stride) dst[i] = f2bf(src[i]);
}

// Pack a fp32 weight matrix [O][Kreal] into bf16 [NT*16][Kpad], zero padded.
__global__ void k_pack_w(const float* __restrict__ w,
                         unsigned short* __restrict__ dst,
                         int O, int Kreal, int Kpad, int NT) {
    int total = NT * 16 * Kpad;
    for (int i = blockIdx.x * blockDim.x + threadIdx.x; i < total;
         i += gridDim.x * blockDim.x) {
        int k = i % Kpad;
        int n = i / Kpad;
        float v = (n < O && k < Kreal) ? w[n * Kreal + k] : 0.0f;
        dst[i] = f2bf(v);
    }
}

// im2col offset table: tab[k] = (c*H + kh)*W + kw for k < Kreal, else -1.
__global__ void k_build_tab(int* __restrict__ tab, int H, int W, int KH, int KW,
                            int Kreal, int Kpad) {
    for (int k = blockIdx.x * blockDim.x + threadIdx.x; k < Kpad;
         k += gridDim.x * blockDim.x) {
        int v = -1;
        if (k < Kreal) {
            int c  = k / (KH * KW);
            int r  = k % (KH * KW);
            int kh = r / KW, kw = r % KW;
            v = (c * H + kh) * W + kw;
        }
        tab[k] = v;
    }
}

// 2x2 max pool: [B][20][24][24] -> [B][20][12][12], bf16.
__global__ void k_pool(const unsigned short* __restrict__ in,
                       unsigned short* __restrict__ out, int B) {
    long long total = (long long)B * 20 * 12 * 12;
    long long i = (long long)blockIdx.x * blockDim.x + threadIdx.x;
    long long stride = (long long)gridDim.x * blockDim.x;
    for (; i < total; i += stride) {
        int ow = (int)(i % 12);
        long long t = i / 12;
        int oh = (int)(t % 12); t /= 12;
        int c  = (int)(t % 20);
        long long b = t / 20;
        const unsigned short* p = in + (((b * 20 + c) * 24) + oh * 2) * 24 + ow * 2;
        float m = bf2f(p[0]);
        m = fmaxf(m, bf2f(p[1]));
        m = fmaxf(m, bf2f(p[24]));
        m = fmaxf(m, bf2f(p[25]));
        out[i] = f2bf(m);
    }
}

// Permuted per-channel 1x1 conv + flatten into padded fc1 activations.
// y3: [B][29][100] bf16 -> a1: [B][1408] bf16 (cols 1400..1407 zero).
__global__ void k_perm(const unsigned short* __restrict__ y3,
                       const float* __restrict__ pw,
                       const float* __restrict__ pb,
                       unsigned short* __restrict__ a1, int B) {
    long long total = (long long)B * 1408;
    long long idx = (long long)blockIdx.x * blockDim.x + threadIdx.x;
    long long stride = (long long)gridDim.x * blockDim.x;
    for (; idx < total; idx += stride) {
        int j = (int)(idx % 1408);
        long long b = idx / 1408;
        unsigned short v = 0;
        if (j < 1400) {
            int i = j / 100, p = j % 100;
            const unsigned short* base = y3 + b * 2900;
            float val = pw[i * 3 + 0] * bf2f(base[p])
                      + pw[i * 3 + 1] * bf2f(base[(2 * i + 1) * 100 + p])
                      + pw[i * 3 + 2] * bf2f(base[(2 * i + 2) * 100 + p])
                      + pb[i];
            v = f2bf(val);
        }
        a1[idx] = v;
    }
}

// --------------------------- WMMA conv kernel ------------------------------
// Implicit im2col GEMM: M = B*OH*OW, N = Cout (padded NT*16), K padded Kpad.
// One wave computes one 16x16 tile; the k->input-offset table lives in LDS.

template<int Cin, int H, int W, int Cout, int KH, int KW, int OH, int OW,
         int Kreal, int Kpad, int NT>
__global__ void k_conv_wmma(const unsigned short* __restrict__ in,
                            const unsigned short* __restrict__ wpk,
                            const float* __restrict__ bias,
                            const int* __restrict__ tab,
                            unsigned short* __restrict__ out, int B) {
    __shared__ int tab_s[Kpad];
    for (int i = threadIdx.x; i < Kpad; i += blockDim.x) tab_s[i] = tab[i];
    __syncthreads();

    const int lane = threadIdx.x & 31;
    const int wave = threadIdx.x >> 5;
    const int hi   = lane >> 4;
    const int lo   = lane & 15;

    const int M = B * OH * OW;
    const int mtiles = (M + 15) >> 4;
    long long tile = (long long)blockIdx.x * (blockDim.x >> 5) + wave;
    if (tile >= (long long)mtiles * NT) return;     // wave-uniform exit
    const int nt = (int)(tile % NT);
    const int mt = (int)(tile / NT);

    // A-fragment source: this lane's output row (im2col row).
    int mrow  = mt * 16 + lo;
    int msafe = (mrow < M) ? mrow : 0;
    int b  = msafe / (OH * OW);
    int r  = msafe % (OH * OW);
    int oh = r / OW, ow = r % OW;
    const unsigned short* ibase =
        in + (long long)b * Cin * H * W + oh * W + ow;

    // B-fragment source: packed weight row n = nt*16 + lo.
    const unsigned short* wrow = wpk + (nt * 16 + lo) * Kpad;

    v8f acc = {};
    #pragma unroll
    for (int kc = 0; kc < Kpad / 32; ++kc) {
        const int kb = kc * 32;
        FragAB a, bw;
        #pragma unroll
        for (int j = 0; j < 8; ++j) {
            // A layout (ISA 7.12.2, 16-bit A 16x32)
            const int koff = kb + ((j < 4) ? 0 : 16) + 8 * hi + (j & 3) * 2;
            #pragma unroll
            for (int h2 = 0; h2 < 2; ++h2) {
                const int off = tab_s[koff + h2];       // ds_load
                unsigned short val = 0;
                if (off >= 0) val = ibase[off];
                a.u[2 * j + h2] = val;
            }
            const int kB = kb + ((j < 4) ? 0 : 16) + 8 * hi + (j & 3) * 2;        // aligned dword pair
            bw.d[j] = *(const unsigned int*)(wrow + kB);
        }
        acc = __builtin_amdgcn_wmma_f32_16x16x32_bf16(
            false, a.v, false, bw.v, (short)0, acc, false, false);
    }

    const int n = nt * 16 + lo;
    const float bn = (n < Cout) ? bias[n] : 0.0f;
    #pragma unroll
    for (int r8 = 0; r8 < 8; ++r8) {
        const int row = mt * 16 + r8 + 8 * hi;          // C/D layout
        if (row < M && n < Cout) {
            const int bb = row / (OH * OW);
            const int rr = row % (OH * OW);
            const int o_h = rr / OW, o_w = rr % OW;
            out[(((long long)bb * Cout + n) * OH + o_h) * OW + o_w] =
                f2bf(acc[r8] + bn);
        }
    }
}

// --------------------------- WMMA GEMM kernel ------------------------------
// One block = one 16-row M-tile; wave w owns N-tile w. The A-tile is staged
// into double-buffered LDS (async-to-LDS on CDNA5) and shared by all waves.

template<int Kpad, int NT, int Nout, int NpadOut, bool F32OUT>
__global__ void k_gemm_wmma(const unsigned short* __restrict__ A,
                            const unsigned short* __restrict__ wpk,
                            const float* __restrict__ bias,
                            unsigned short* __restrict__ outb,
                            float* __restrict__ outf, int M) {
    __shared__ unsigned short atile[2][16 * 32];

    const int lane = threadIdx.x & 31;
    const int nt   = threadIdx.x >> 5;      // wave id == N-tile
    const int hi   = lane >> 4;
    const int lo   = lane & 15;
    const int mt   = blockIdx.x;
    const int mbase = mt * 16;

    // Stage K-chunk kc of the A-tile into LDS buffer `buf`.
    auto stage = [&](int kc, int buf) {
        for (int i = threadIdx.x; i < 256; i += NT * 32) {
            const int row = i >> 4;          // 0..15
            const int dw  = i & 15;          // dword within 32-element chunk
            int m = mbase + row; if (m >= M) m = M - 1;
            const unsigned short* src = A + (long long)m * Kpad + kc * 32 + dw * 2;
            unsigned short* dst = &atile[buf][row * 32 + dw * 2];
#if USE_ASYNC_LDS
            __builtin_amdgcn_global_load_async_to_lds_b32(
                (as1_int*)(unsigned long long)src,
                (as3_int*)(unsigned int)(unsigned long long)dst, 0, 0);
#else
            *(unsigned int*)dst = *(const unsigned int*)src;
#endif
        }
    };

    const unsigned short* wrow = wpk + (nt * 16 + lo) * Kpad;
    constexpr int NC = Kpad / 32;

    stage(0, 0);                             // prologue

    v8f acc = {};
    for (int kc = 0; kc < NC; ++kc) {
#if USE_ASYNC_LDS
        __builtin_amdgcn_s_wait_asynccnt(0);
#endif
        __syncthreads();                     // chunk kc resident & visible
        if (kc + 1 < NC) stage(kc + 1, (kc + 1) & 1);

        const unsigned short* abuf = atile[kc & 1];
        FragAB a, bw;
        #pragma unroll
        for (int j = 0; j < 8; ++j) {
            const int kA = ((j < 4) ? 0 : 16) + 8 * hi + (j & 3) * 2;
            a.d[j]  = *(const unsigned int*)(abuf + lo * 32 + kA);   // ds_load
            const int kB = kc * 32 + ((j < 4) ? 0 : 16) + 8 * hi + (j & 3) * 2;
            bw.d[j] = *(const unsigned int*)(wrow + kB);
        }
        acc = __builtin_amdgcn_wmma_f32_16x16x32_bf16(
            false, a.v, false, bw.v, (short)0, acc, false, false);
    }

    const int n = nt * 16 + lo;
    const float bn = (n < Nout) ? bias[n] : 0.0f;
    #pragma unroll
    for (int r8 = 0; r8 < 8; ++r8) {
        const int row = mbase + r8 + 8 * hi;
        if (row < M) {
            const float v = acc[r8] + bn;
            if (F32OUT) {
                if (n < Nout) outf[(long long)row * Nout + n] = v;
            } else {
                if (n < NpadOut) outb[(long long)row * NpadOut + n] = f2bf(v);
            }
        }
    }
}

// ------------------------------- launch ------------------------------------

extern "C" void kernel_launch(void* const* d_in, const int* in_sizes, int n_in,
                              void* d_out, int out_size, void* d_ws, size_t ws_size,
                              hipStream_t stream) {
    const float* x   = (const float*)d_in[0];
    const float* c1w = (const float*)d_in[1];  const float* c1b = (const float*)d_in[2];
    const float* c2w = (const float*)d_in[3];  const float* c2b = (const float*)d_in[4];
    const float* c3w = (const float*)d_in[5];  const float* c3b = (const float*)d_in[6];
    const float* pw  = (const float*)d_in[7];  const float* pb  = (const float*)d_in[8];
    const float* f1w = (const float*)d_in[9];  const float* f1b = (const float*)d_in[10];
    const float* f2w = (const float*)d_in[11]; const float* f2b = (const float*)d_in[12];
    const float* f3w = (const float*)d_in[13]; const float* f3b = (const float*)d_in[14];
    float* out = (float*)d_out;

    const int B = in_sizes[0] / (3 * 32 * 32);   // 8192

    // ---- workspace layout ----
    char* wsb = (char*)d_ws;
    unsigned short* w1p = (unsigned short*)(wsb);                       // 16*96
    unsigned short* w2p = w1p + 16 * 96;                                // 32*256
    unsigned short* w3p = w2p + 32 * 256;                               // 32*192
    unsigned short* f1p = w3p + 32 * 192;                               // 128*1408
    unsigned short* f2p = f1p + 128 * 1408;                             // 96*128
    unsigned short* f3p = f2p + 96 * 128;                               // 16*96
    int* tb1 = (int*)(((size_t)(f3p + 16 * 96) + 255) & ~(size_t)255);  // 96 ints
    int* tb2 = tb1 + 96;                                                // 256 ints
    int* tb3 = tb2 + 256;                                               // 192 ints
    size_t woff = (size_t)((char*)(tb3 + 192) - wsb);
    woff = (woff + 255) & ~(size_t)255;

    // Ping-pong arenas for activations (bf16).
    const size_t sz_xh = (size_t)B * 3 * 32 * 32 * 2;
    const size_t sz_y2 = (size_t)B * 20 * 24 * 24 * 2;
    const size_t sz_y3 = (size_t)B * 29 * 10 * 10 * 2;
    size_t a0 = sz_xh;  if (sz_y2 > a0) a0 = sz_y2;  if (sz_y3 > a0) a0 = sz_y3;
    unsigned short* arena0 = (unsigned short*)(wsb + woff);
    unsigned short* arena1 = (unsigned short*)(wsb + woff + ((a0 + 255) & ~(size_t)255));

    unsigned short* xh  = arena0;   // conv1 input (bf16 of x)
    unsigned short* y1  = arena1;   // conv1 out [B][10][28][28]
    unsigned short* y2  = arena0;   // conv2 out [B][20][24][24]
    unsigned short* y2p = arena1;   // pooled   [B][20][12][12]
    unsigned short* y3  = arena0;   // conv3 out [B][29][10][10]
    unsigned short* a1  = arena1;   // fc1 acts [B][1408]
    unsigned short* h1  = arena0;   // fc1 out  [B][128]
    unsigned short* h2  = arena1;   // fc2 out  [B][96]

    // ---- 1) pack weights + build im2col tables ----
    k_pack_w<<<64, 256, 0, stream>>>(c1w, w1p, 10,  75,   96, 1);
    k_pack_w<<<64, 256, 0, stream>>>(c2w, w2p, 20,  250,  256, 2);
    k_pack_w<<<64, 256, 0, stream>>>(c3w, w3p, 29,  180,  192, 2);
    k_pack_w<<<256, 256, 0, stream>>>(f1w, f1p, 120, 1400, 1408, 8);
    k_pack_w<<<64, 256, 0, stream>>>(f2w, f2p, 84,  120,  128, 6);
    k_pack_w<<<64, 256, 0, stream>>>(f3w, f3p, 10,  84,   96, 1);
    k_build_tab<<<1, 256, 0, stream>>>(tb1, 32, 32, 5, 5, 75, 96);
    k_build_tab<<<1, 256, 0, stream>>>(tb2, 28, 28, 5, 5, 250, 256);
    k_build_tab<<<1, 256, 0, stream>>>(tb3, 12, 12, 3, 3, 180, 192);

    // ---- 2) input fp32 -> bf16 ----
    k_f32_to_bf16<<<4096, 256, 0, stream>>>(x, xh, (long long)B * 3 * 32 * 32);

    // ---- 3) conv1: 3x32x32 -> 10x28x28 (K 75->96, NT=1) ----
    {
        long long waves = ((long long)(B * 28 * 28 + 15) / 16) * 1;
        int blocks = (int)((waves + 7) / 8);
        k_conv_wmma<3, 32, 32, 10, 5, 5, 28, 28, 75, 96, 1>
            <<<blocks, 256, 0, stream>>>(xh, w1p, c1b, tb1, y1, B);
    }
    // ---- 4) conv2: 10x28x28 -> 20x24x24 (K 250->256, NT=2) ----
    {
        long long waves = ((long long)(B * 24 * 24 + 15) / 16) * 2;
        int blocks = (int)((waves + 7) / 8);
        k_conv_wmma<10, 28, 28, 20, 5, 5, 24, 24, 250, 256, 2>
            <<<blocks, 256, 0, stream>>>(y1, w2p, c2b, tb2, y2, B);
    }
    // ---- 5) 2x2 max pool ----
    k_pool<<<4096, 256, 0, stream>>>(y2, y2p, B);

    // ---- 6) conv3: 20x12x12 -> 29x10x10 (K 180->192, NT=2) ----
    {
        long long waves = ((long long)(B * 10 * 10 + 15) / 16) * 2;
        int blocks = (int)((waves + 7) / 8);
        k_conv_wmma<20, 12, 12, 29, 3, 3, 10, 10, 180, 192, 2>
            <<<blocks, 256, 0, stream>>>(y2p, w3p, c3b, tb3, y3, B);
    }
    // ---- 7) permuted 1x1 conv + flatten to [B][1408] ----
    k_perm<<<4096, 256, 0, stream>>>(y3, pw, pb, a1, B);

    // ---- 8) fc1: [B,1408] x [120,1400]^T -> [B,128] bf16 ----
    {
        int blocks = (B + 15) / 16;          // 512 M-tiles, 8 N-tiles/block
        k_gemm_wmma<1408, 8, 120, 128, false>
            <<<blocks, 8 * 32, 0, stream>>>(a1, f1p, f1b, h1, nullptr, B);
    }
    // ---- 9) fc2: [B,128] x [84,120]^T -> [B,96] bf16 ----
    {
        int blocks = (B + 15) / 16;          // 6 N-tiles/block
        k_gemm_wmma<128, 6, 84, 96, false>
            <<<blocks, 6 * 32, 0, stream>>>(h1, f2p, f2b, h2, nullptr, B);
    }
    // ---- 10) fc3: [B,96] x [10,84]^T -> [B,10] fp32 (d_out) ----
    {
        int blocks = (B + 15) / 16;          // 1 N-tile/block
        k_gemm_wmma<96, 1, 10, 10, true>
            <<<blocks, 1 * 32, 0, stream>>>(h2, f3p, f3b, nullptr, out, B);
    }
    (void)n_in; (void)out_size; (void)ws_size;
}
